// SRGNNEncoder_65730179497991
// MI455X (gfx1250) — compile-verified, matrix operand order checked by reading, not run on hardware
//
#include <hip/hip_runtime.h>
#include <hip/hip_bf16.h>

typedef __attribute__((ext_vector_type(16))) _Float16 v16h;
typedef __attribute__((ext_vector_type(8)))  _Float16 v8h;
typedef __attribute__((ext_vector_type(8)))  float    v8f;

#define Bsz 512
#define Nn  64
#define Lsz 64
#define Dd  256

__device__ __forceinline__ float sigmoidf_(float x) {
    return 1.0f / (1.0f + __expf(-x));
}

// Load a 16-element f16 fragment: 8 contiguous at p, 8 contiguous at p+16.
// Both halves are 16-byte aligned -> two global_load_b128.
__device__ __forceinline__ v16h load_frag16(const _Float16* __restrict__ p) {
    const v8h lo = *reinterpret_cast<const v8h*>(p);
    const v8h hi = *reinterpret_cast<const v8h*>(p + 16);
    v16h r;
#pragma unroll
    for (int e = 0; e < 8; ++e) { r[e] = lo[e]; r[8 + e] = hi[e]; }
    return r;
}

// ---------------------------------------------------------------------------
// GEMM: C[M,N] = act(A[M,K] @ W[K,N] + bias[N])
//   A:  f16 row-major [M,K]
//   Wt: f16 TRANSPOSED weights [N,K] (row n = column n of W)
// Outputs (any may be null):
//   C32  f32 [M,N], C16 f16 [M,N], C16T f16 [N,M]
// Requirements: K % 32 == 0, M % 16 == 0, N % 128 == 0.
// blockDim = 128 (4 waves); each wave computes a 16(M) x 32(N) strip.
// grid = (N/128, M/16). act: 0 = none, 1 = relu.
// ---------------------------------------------------------------------------
__global__ void gemm_wmma_f16(const _Float16* __restrict__ A,
                              const _Float16* __restrict__ Wt,
                              const float* __restrict__ bias,
                              float* __restrict__ C32,
                              _Float16* __restrict__ C16,
                              _Float16* __restrict__ C16T,
                              int M, int K, int N, int act)
{
    const int wave = threadIdx.x >> 5;
    const int lane = threadIdx.x & 31;
    const int half = lane >> 4;          // K sub-range select (ISA 7.12.2)
    const int l16  = lane & 15;
    const int nt   = (blockIdx.x * 4 + wave) * 32;   // 32-wide N strip
    const int rowA = blockIdx.y * 16 + l16;
    const int col0 = nt + l16;
    const int col1 = nt + 16 + l16;

    v8f acc0 = {}, acc1 = {};
    for (int kb = 0; kb < K; kb += 32) {
        const v16h a  = load_frag16(A  + (size_t)rowA * K + kb + half * 8);
        const v16h b0 = load_frag16(Wt + (size_t)col0 * K + kb + half * 8);
        const v16h b1 = load_frag16(Wt + (size_t)col1 * K + kb + half * 8);
        acc0 = __builtin_amdgcn_wmma_f32_16x16x32_f16(
            false, a, false, b0, (short)0, acc0, false, false);
        acc1 = __builtin_amdgcn_wmma_f32_16x16x32_f16(
            false, a, false, b1, (short)0, acc1, false, false);
    }

    const float bv0 = bias ? bias[col0] : 0.0f;
    const float bv1 = bias ? bias[col1] : 0.0f;
#pragma unroll
    for (int r = 0; r < 8; ++r) {
        const int m = blockIdx.y * 16 + r + half * 8;  // C/D: M = r + 8*half
        float v0 = acc0[r] + bv0;
        float v1 = acc1[r] + bv1;
        if (act == 1) { v0 = fmaxf(v0, 0.0f); v1 = fmaxf(v1, 0.0f); }
        if (C32) {
            C32[(size_t)m * N + col0] = v0;
            C32[(size_t)m * N + col1] = v1;
        }
        if (C16) {
            C16[(size_t)m * N + col0] = (_Float16)v0;
            C16[(size_t)m * N + col1] = (_Float16)v1;
        }
        if (C16T) {
            C16T[(size_t)col0 * M + m] = (_Float16)v0;
            C16T[(size_t)col1 * M + m] = (_Float16)v1;
        }
    }
}

// ---------------------------------------------------------------------------
// Batched adjacency GEMM:
//   Out16[b, m, ooff+n] = Adj16[b][m, aoff+k] @ XT[n, b*Nn+k] + bias[n]
//   Adj16: f16 [B, Nn, 2*Nn];  XT: f16 transposed activations [Dd, B*Nn]
//   Out16: f16 [B*Nn, 2*Dd]
// M = K = Nn = 64, N = Dd = 256.
// blockDim = 128 (4 waves, 16x32 strip each). grid = (Dd/128, Nn/16, B).
// ---------------------------------------------------------------------------
__global__ void bgemm_adj_wmma(const _Float16* __restrict__ Adj16, int aoff,
                               const _Float16* __restrict__ XT,
                               const float* __restrict__ bias,
                               _Float16* __restrict__ Out16, int ooff)
{
    const int b    = blockIdx.z;
    const int wave = threadIdx.x >> 5;
    const int lane = threadIdx.x & 31;
    const int half = lane >> 4;
    const int l16  = lane & 15;
    const int nt   = (blockIdx.x * 4 + wave) * 32;
    const int rowA = blockIdx.y * 16 + l16;
    const int col0 = nt + l16;
    const int col1 = nt + 16 + l16;

    const size_t BN = (size_t)Bsz * Nn;
    const _Float16* Ab = Adj16 + (size_t)b * Nn * (2 * Nn) + aoff;

    v8f acc0 = {}, acc1 = {};
    for (int kb = 0; kb < Nn; kb += 32) {
        const v16h a  = load_frag16(Ab + (size_t)rowA * (2 * Nn) + kb + half * 8);
        const v16h b0 = load_frag16(XT + (size_t)col0 * BN + (size_t)b * Nn + kb + half * 8);
        const v16h b1 = load_frag16(XT + (size_t)col1 * BN + (size_t)b * Nn + kb + half * 8);
        acc0 = __builtin_amdgcn_wmma_f32_16x16x32_f16(
            false, a, false, b0, (short)0, acc0, false, false);
        acc1 = __builtin_amdgcn_wmma_f32_16x16x32_f16(
            false, a, false, b1, (short)0, acc1, false, false);
    }

    const float bv0 = bias[col0];
    const float bv1 = bias[col1];
    _Float16* Ob = Out16 + (size_t)b * Nn * (2 * Dd) + ooff;
#pragma unroll
    for (int r = 0; r < 8; ++r) {
        const int m = blockIdx.y * 16 + r + half * 8;
        Ob[(size_t)m * (2 * Dd) + col0] = (_Float16)(acc0[r] + bv0);
        Ob[(size_t)m * (2 * Dd) + col1] = (_Float16)(acc1[r] + bv1);
    }
}

// ---------------------------------------------------------------------------
// Conversion / elementwise / gather kernels
// ---------------------------------------------------------------------------
__global__ void convert_f16(const float* __restrict__ x, _Float16* __restrict__ y)
{
    const size_t idx = (size_t)blockIdx.x * blockDim.x + threadIdx.x;
    y[idx] = (_Float16)x[idx];
}

// Wt[n*K + k] = (f16) W[k*N + n]  (weights stored [in=K, out=N] in the ref)
__global__ void convert_w_t(const float* __restrict__ W, _Float16* __restrict__ Wt,
                            int K, int N)
{
    const size_t idx = (size_t)blockIdx.x * blockDim.x + threadIdx.x; // over N*K
    const int n = (int)(idx / K);
    const int k = (int)(idx % K);
    Wt[idx] = (_Float16)W[(size_t)k * N + n];
}

__global__ void gather_embed(const float* __restrict__ emb,
                             const int* __restrict__ items,
                             float* __restrict__ h32,
                             _Float16* __restrict__ h16)
{
    const size_t idx = (size_t)blockIdx.x * blockDim.x + threadIdx.x; // B*Nn*Dd
    const size_t row = idx / Dd;
    const int    d   = (int)(idx % Dd);
    const float  v   = emb[(size_t)items[row] * Dd + d];
    h32[idx] = v;
    h16[idx] = (_Float16)v;
}

__global__ void gru_gate(const float* __restrict__ gi,
                         const float* __restrict__ gh,
                         const float* __restrict__ h,
                         _Float16* __restrict__ hnew16)
{
    const size_t idx = (size_t)blockIdx.x * blockDim.x + threadIdx.x; // B*Nn*Dd
    const size_t row = idx / Dd;
    const int    d   = (int)(idx % Dd);
    const float* gir = gi + row * (3 * Dd);
    const float* ghr = gh + row * (3 * Dd);
    const float r  = sigmoidf_(gir[d] + ghr[d]);
    const float g  = sigmoidf_(gir[Dd + d] + ghr[Dd + d]);
    const float nw = tanhf(gir[2 * Dd + d] + r * ghr[2 * Dd + d]);
    const float hv = h[idx];
    hnew16[idx] = (_Float16)(nw + g * (hv - nw));
}

__global__ void gather_seq(const _Float16* __restrict__ hid16,
                           const int* __restrict__ alias,
                           _Float16* __restrict__ seq16)
{
    const size_t idx = (size_t)blockIdx.x * blockDim.x + threadIdx.x; // B*L*Dd
    const size_t bl  = idx / Dd;
    const int    d   = (int)(idx % Dd);
    const size_t b   = bl / Lsz;
    seq16[idx] = hid16[((size_t)b * Nn + alias[bl]) * Dd + d];
}

// ht16[b,:] = hidden[b, alias[b, sum(mask)-1], :]; also fills x16[:, Dd:2Dd]
__global__ void compute_ht(const _Float16* __restrict__ hid16,
                           const int* __restrict__ alias,
                           const int* __restrict__ mask,
                           _Float16* __restrict__ ht16,
                           _Float16* __restrict__ x16)
{
    const int b = blockIdx.x;
    const int d = threadIdx.x;        // 256
    __shared__ int lastS;
    if (d == 0) {
        int s = 0;
        for (int l = 0; l < Lsz; ++l) s += mask[b * Lsz + l];
        lastS = s - 1;
    }
    __syncthreads();
    const int node = alias[b * Lsz + lastS];
    const _Float16 v = hid16[((size_t)b * Nn + node) * Dd + d];
    ht16[(size_t)b * Dd + d] = v;
    x16[(size_t)b * (2 * Dd) + Dd + d] = v;
}

// alpha[b,l] = sum_d sigmoid(q1[b,d] + q2[b,l,d]) * W3[d]
__global__ void attn_alpha(const float* __restrict__ q1,
                           const float* __restrict__ q2,
                           const float* __restrict__ W3,
                           float* __restrict__ alpha)
{
    const int bl = blockIdx.x;        // B*L blocks
    const int d  = threadIdx.x;       // 256
    const int b  = bl / Lsz;
    float v = sigmoidf_(q1[(size_t)b * Dd + d] + q2[(size_t)bl * Dd + d]) * W3[d];
    __shared__ float red[Dd];
    red[d] = v;
    __syncthreads();
    for (int s = Dd / 2; s > 0; s >>= 1) {
        if (d < s) red[d] += red[d + s];
        __syncthreads();
    }
    if (d == 0) alpha[bl] = red[0];
}

// x16[b, d] = sum_l alpha[b,l] * seq[b,l,d] * mask[b,l]
__global__ void attn_sum(const float* __restrict__ alpha,
                         const _Float16* __restrict__ seq16,
                         const int* __restrict__ mask,
                         _Float16* __restrict__ x16)
{
    const int b = blockIdx.x;
    const int d = threadIdx.x;        // 256
    float s = 0.0f;
    for (int l = 0; l < Lsz; ++l) {
        const int bl = b * Lsz + l;
        s += alpha[bl] * (float)seq16[(size_t)bl * Dd + d] * (float)mask[bl];
    }
    x16[(size_t)b * (2 * Dd) + d] = (_Float16)s;
}

// ---------------------------------------------------------------------------
// Host launcher
// ---------------------------------------------------------------------------
extern "C" void kernel_launch(void* const* d_in, const int* in_sizes, int n_in,
                              void* d_out, int out_size, void* d_ws, size_t ws_size,
                              hipStream_t stream) {
    (void)in_sizes; (void)n_in; (void)out_size; (void)ws_size;

    const int*   alias  = (const int*)  d_in[0];
    const float* Adj    = (const float*)d_in[1];
    const int*   items  = (const int*)  d_in[2];
    const int*   mask   = (const int*)  d_in[3];
    const float* emb    = (const float*)d_in[4];
    const float* W_ein  = (const float*)d_in[5];
    const float* b_ein  = (const float*)d_in[6];
    const float* W_eout = (const float*)d_in[7];
    const float* b_eout = (const float*)d_in[8];
    const float* b_iah  = (const float*)d_in[9];
    const float* b_oah  = (const float*)d_in[10];
    const float* W_i    = (const float*)d_in[11];
    const float* b_i    = (const float*)d_in[12];
    const float* W_h    = (const float*)d_in[13];
    const float* b_h    = (const float*)d_in[14];
    const float* W1     = (const float*)d_in[15];
    const float* b1     = (const float*)d_in[16];
    const float* W2     = (const float*)d_in[17];
    const float* b2     = (const float*)d_in[18];
    const float* W3     = (const float*)d_in[19];
    const float* Wn1    = (const float*)d_in[20];
    const float* bn1    = (const float*)d_in[21];
    const float* Wn2    = (const float*)d_in[22];
    const float* bn2    = (const float*)d_in[23];
    float* out = (float*)d_out;

    const size_t BN = (size_t)Bsz * Nn;   // 32768 rows

    // ---- workspace carve (bytes), with lifetime-based aliasing ----
    char* cur = (char*)d_ws;
    auto take = [&](size_t bytes) { char* p = cur; cur += (bytes + 255) & ~(size_t)255; return p; };

    float*    h32      = (float*)   take(BN * Dd * 4);            // live: whole
    _Float16* h16      = (_Float16*)take(BN * Dd * 2);            // live: whole
    _Float16* adj16    = (_Float16*)take((size_t)Bsz * Nn * 2 * Nn * 2);
    _Float16* t0T      = (_Float16*)take((size_t)Dd * BN * 2);    // later: hnew16
    _Float16* inputs16 = (_Float16*)take(BN * 2 * Dd * 2);
    // transposed f16 weights (live whole launch, ~2.1 MB)
    _Float16* WeinT  = (_Float16*)take((size_t)Dd * Dd * 2);
    _Float16* WeoutT = (_Float16*)take((size_t)Dd * Dd * 2);
    _Float16* WiT    = (_Float16*)take((size_t)3 * Dd * 2 * Dd * 2);
    _Float16* WhT    = (_Float16*)take((size_t)3 * Dd * Dd * 2);
    _Float16* W1T    = (_Float16*)take((size_t)Dd * Dd * 2);
    _Float16* W2T    = (_Float16*)take((size_t)Dd * Dd * 2);
    _Float16* Wn1T   = (_Float16*)take((size_t)Dd * 2 * Dd * 2);
    _Float16* Wn2T   = (_Float16*)take((size_t)Dd * Dd * 2);
    float*    gi32   = (float*)take(BN * 3 * Dd * 4);             // later: seq16,q2,...
    float*    gh32   = (float*)take(BN * 3 * Dd * 4);

    // aliases (producers/consumers strictly ordered on one stream)
    _Float16* hnew16 = t0T;                       // t0T dead after step 5
    char* giR = (char*)gi32;                      // gi32 dead after gating
    _Float16* seq16  = (_Float16*)giR;                            // BN*Dd f16
    float*    q2     = (float*)(giR + BN * Dd * 2);               // BN*Dd f32
    float*    q1     = (float*)(giR + BN * Dd * 2 + BN * Dd * 4); // B*Dd f32
    _Float16* ht16   = (_Float16*)((char*)q1 + (size_t)Bsz * Dd * 4);
    float*    alphav = (float*)((char*)ht16 + (size_t)Bsz * Dd * 2);
    _Float16* x16    = (_Float16*)((char*)alphav + (size_t)Bsz * Lsz * 4);
    _Float16* tmp16  = (_Float16*)((char*)x16 + (size_t)Bsz * 2 * Dd * 2);

    const dim3 blk128(128);
    const int ewBlocks = (int)(BN * Dd / 256);    // 32768

    // 0. one-time (per launch) f16 conversions
    convert_f16<<<(int)((size_t)Bsz * Nn * 2 * Nn / 256), 256, 0, stream>>>(Adj, adj16);
    convert_w_t<<<Dd * Dd / 256, 256, 0, stream>>>(W_ein,  WeinT,  Dd,     Dd);
    convert_w_t<<<Dd * Dd / 256, 256, 0, stream>>>(W_eout, WeoutT, Dd,     Dd);
    convert_w_t<<<2 * Dd * 3 * Dd / 256, 256, 0, stream>>>(W_i, WiT, 2 * Dd, 3 * Dd);
    convert_w_t<<<Dd * 3 * Dd / 256, 256, 0, stream>>>(W_h, WhT, Dd, 3 * Dd);
    convert_w_t<<<Dd * Dd / 256, 256, 0, stream>>>(W1,  W1T,  Dd,     Dd);
    convert_w_t<<<Dd * Dd / 256, 256, 0, stream>>>(W2,  W2T,  Dd,     Dd);
    convert_w_t<<<2 * Dd * Dd / 256, 256, 0, stream>>>(Wn1, Wn1T, 2 * Dd, Dd);
    convert_w_t<<<Dd * Dd / 256, 256, 0, stream>>>(Wn2, Wn2T, Dd, Dd);

    // 1. hidden = emb[items]  (f32 + f16 copies)
    gather_embed<<<ewBlocks, 256, 0, stream>>>(emb, items, h32, h16);

    // 2. t0T = (h @ W_ein + b_ein)^T   [Dd, BN] f16
    gemm_wmma_f16<<<dim3(Dd / 128, (int)(BN / 16)), blk128, 0, stream>>>(
        h16, WeinT, b_ein, nullptr, nullptr, t0T, (int)BN, Dd, Dd, 0);
    // 3. inputs[:, :, 0:256] = A_in @ t0 + b_iah
    bgemm_adj_wmma<<<dim3(Dd / 128, Nn / 16, Bsz), blk128, 0, stream>>>(
        adj16, 0, t0T, b_iah, inputs16, 0);
    // 4. t0T = (h @ W_eout + b_eout)^T
    gemm_wmma_f16<<<dim3(Dd / 128, (int)(BN / 16)), blk128, 0, stream>>>(
        h16, WeoutT, b_eout, nullptr, nullptr, t0T, (int)BN, Dd, Dd, 0);
    // 5. inputs[:, :, 256:512] = A_out @ t0 + b_oah
    bgemm_adj_wmma<<<dim3(Dd / 128, Nn / 16, Bsz), blk128, 0, stream>>>(
        adj16, Nn, t0T, b_oah, inputs16, Dd);

    // 6. gi = inputs @ W_i + b_i   [BN, 768] f32
    gemm_wmma_f16<<<dim3(3 * Dd / 128, (int)(BN / 16)), blk128, 0, stream>>>(
        inputs16, WiT, b_i, gi32, nullptr, nullptr, (int)BN, 2 * Dd, 3 * Dd, 0);
    // 7. gh = h @ W_h + b_h        [BN, 768] f32
    gemm_wmma_f16<<<dim3(3 * Dd / 128, (int)(BN / 16)), blk128, 0, stream>>>(
        h16, WhT, b_h, gh32, nullptr, nullptr, (int)BN, Dd, 3 * Dd, 0);

    // 8. GRU gating -> hnew16 (aliases t0T)
    gru_gate<<<ewBlocks, 256, 0, stream>>>(gi32, gh32, h32, hnew16);

    // 9. seq gather; 10. ht (fills x16[:, 256:512])
    gather_seq<<<ewBlocks, 256, 0, stream>>>(hnew16, alias, seq16);
    compute_ht<<<Bsz, Dd, 0, stream>>>(hnew16, alias, mask, ht16, x16);

    // 11. q1 = ht @ W1 + b1; 12. q2 = seq @ W2 + b2
    gemm_wmma_f16<<<dim3(Dd / 128, Bsz / 16), blk128, 0, stream>>>(
        ht16, W1T, b1, q1, nullptr, nullptr, Bsz, Dd, Dd, 0);
    gemm_wmma_f16<<<dim3(Dd / 128, (int)(BN / 16)), blk128, 0, stream>>>(
        seq16, W2T, b2, q2, nullptr, nullptr, (int)BN, Dd, Dd, 0);

    // 13. alpha = sigmoid(q1 + q2) @ W3; 14. a -> x16[:, 0:256]
    attn_alpha<<<Bsz * Lsz, Dd, 0, stream>>>(q1, q2, W3, alphav);
    attn_sum<<<Bsz, Dd, 0, stream>>>(alphav, seq16, mask, x16);

    // 15. tmp = relu(x @ Wn1 + bn1) -> f16; 16. out = tmp @ Wn2 + bn2 -> f32
    gemm_wmma_f16<<<dim3(Dd / 128, Bsz / 16), blk128, 0, stream>>>(
        x16, Wn1T, bn1, nullptr, tmp16, nullptr, Bsz, 2 * Dd, Dd, 1);
    gemm_wmma_f16<<<dim3(Dd / 128, Bsz / 16), blk128, 0, stream>>>(
        tmp16, Wn2T, bn2, out, nullptr, nullptr, Bsz, Dd, Dd, 0);
}
